// TransformerBlock_11802570129582
// MI455X (gfx1250) — matrix operand order repaired and checked by reading, not compiled, not run on hardware
//
#include <hip/hip_runtime.h>
#include <hip/hip_bf16.h>

// ---------------------------------------------------------------------------
// Transformer block for MI455X (gfx1250): bf16 WMMA (v_wmma_f32_16x16x32_bf16)
// everywhere, fp32 accumulation, flash-attention with online softmax.
// - Weights converted fp32->bf16 + transposed once; GEMM inner loop is pure
//   b128 LDS staging + WMMA, BK=64, register double-buffered.
// - Softmax row-max via v_permlane16 (VALU, no DS waits); row-sum via an
//   extra WMMA against an all-ones B fragment.
// ---------------------------------------------------------------------------

typedef __attribute__((ext_vector_type(16))) __bf16 v16bf;
typedef __attribute__((ext_vector_type(8)))  float  v8f;

#define DIM     1024
#define HEADS   16
#define HDIM    64
#define HIDDEN  4096
#define SEQ     2048
#define BATCH   4
#define MTOT    (BATCH * SEQ)      // 8192 tokens
#define EPS     1e-5f
#define SM_SCALE 0.03125f                      // DIM^-0.5 = 1/32 (exact)
#define SM_SCALE_LOG2E 0.04508422772467139f    // SM_SCALE * log2(e)

// ---- bf16 helpers ---------------------------------------------------------
__device__ __forceinline__ unsigned short f2bf(float f) {
    unsigned int u = __float_as_uint(f);
    u += 0x7FFFu + ((u >> 16) & 1u);            // round-to-nearest-even
    return (unsigned short)(u >> 16);
}

__device__ __forceinline__ v8f wmma_bf16(v16bf a, v16bf b, v8f c) {
    return __builtin_amdgcn_wmma_f32_16x16x32_bf16(
        /*neg_a=*/false, a, /*neg_b=*/false, b,
        /*c_mod=*/(short)0, c, /*reuse_a=*/false, /*reuse_b=*/false);
}

// max-reduce across each 16-lane group (lanes 0-15 and 16-31 independently)
#if __has_builtin(__builtin_amdgcn_permlane16)
__device__ __forceinline__ float grp16_max(float x) {
    unsigned u;
    u = __builtin_amdgcn_permlane16(__float_as_uint(x), __float_as_uint(x),
                                    0x67452301u, 0xEFCDAB89u, false, false); // xor1
    x = fmaxf(x, __uint_as_float(u));
    u = __builtin_amdgcn_permlane16(__float_as_uint(x), __float_as_uint(x),
                                    0x54761032u, 0xDCFE98BAu, false, false); // xor2
    x = fmaxf(x, __uint_as_float(u));
    u = __builtin_amdgcn_permlane16(__float_as_uint(x), __float_as_uint(x),
                                    0x32107654u, 0xBA98FEDCu, false, false); // xor4
    x = fmaxf(x, __uint_as_float(u));
    u = __builtin_amdgcn_permlane16(__float_as_uint(x), __float_as_uint(x),
                                    0xFEDCBA98u, 0x76543210u, false, false); // xor8
    x = fmaxf(x, __uint_as_float(u));
    return x;
}
#else
__device__ __forceinline__ float grp16_max(float x) {
#pragma unroll
    for (int m = 8; m >= 1; m >>= 1) x = fmaxf(x, __shfl_xor(x, m, 32));
    return x;
}
#endif

// A fragment (16x32, 16-bit): lane L holds row = row_base + L%16.
// dword v : K = 16*(L/16) + 2v (contiguous, matches load_b_frag).
__device__ __forceinline__ v16bf load_a_frag(const unsigned short* base,
                                             int row_base, int pitch, int col) {
    const int lane = threadIdx.x & 31;
    const int g = lane >> 4;
    const unsigned short* p = base + (size_t)(row_base + (lane & 15)) * pitch + col;
    union { v16bf v; unsigned int u[8]; } f;
#pragma unroll
    for (int v = 0; v < 8; ++v) {
        f.u[v]     = *(const unsigned int*)(p + 16 * g + 2 * v);
    }
    return f.v;
}

// B fragment (32x16, 16-bit): lane L holds col n = n_base + L%16,
// K = 16*(L/16) + j (j = 0..15 contiguous).  Source stores B transposed: [n][K].
__device__ __forceinline__ v16bf load_b_frag(const unsigned short* base,
                                             int n_base, int pitch, int k_off) {
    const int lane = threadIdx.x & 31;
    const unsigned short* p = base + (size_t)(n_base + (lane & 15)) * pitch
                                   + k_off + 16 * (lane >> 4);
    union { v16bf v; unsigned int u[8]; } f;
#pragma unroll
    for (int v = 0; v < 8; ++v) f.u[v] = *(const unsigned int*)(p + 2 * v);
    return f.v;
}

// ---------------------------------------------------------------------------
// Weight convert + transpose: W[K][N] fp32  ->  Wt[N][K] bf16  (one-time)
// ---------------------------------------------------------------------------
__global__ void __launch_bounds__(256)
wt_kernel(const float* __restrict__ W, unsigned short* __restrict__ Wt,
          int K, int N) {
    __shared__ unsigned short t[64][72];
    const int k0 = blockIdx.y * 64, n0 = blockIdx.x * 64;
    const int nx = threadIdx.x & 63, ky = threadIdx.x >> 6;   // 64 x 4
#pragma unroll
    for (int i = 0; i < 16; ++i) {
        const int k = ky + i * 4;
        t[nx][k] = f2bf(W[(size_t)(k0 + k) * N + n0 + nx]);   // coalesced read
    }
    __syncthreads();
    const int kx = threadIdx.x & 63, ny = threadIdx.x >> 6;
#pragma unroll
    for (int i = 0; i < 16; ++i) {
        const int n = ny + i * 4;
        Wt[(size_t)(n0 + n) * K + k0 + kx] = t[n][kx];        // coalesced write
    }
}

// ---------------------------------------------------------------------------
// LayerNorm (unbiased var, std+eps, matches reference) : fp32 in -> bf16 out
// ---------------------------------------------------------------------------
__global__ void __launch_bounds__(256)
ln_kernel(const float* __restrict__ x, const float* __restrict__ gamma,
          const float* __restrict__ beta, unsigned short* __restrict__ out) {
    __shared__ float red[8];
    const int row = blockIdx.x;
    const float* xr = x + (size_t)row * DIM;
    float v[4];
#pragma unroll
    for (int i = 0; i < 4; ++i) v[i] = xr[threadIdx.x + i * 256];

    float s = v[0] + v[1] + v[2] + v[3];
#pragma unroll
    for (int m = 16; m >= 1; m >>= 1) s += __shfl_xor(s, m, 32);
    if ((threadIdx.x & 31) == 0) red[threadIdx.x >> 5] = s;
    __syncthreads();
    float tot = 0.f;
#pragma unroll
    for (int i = 0; i < 8; ++i) tot += red[i];
    const float mean = tot * (1.0f / DIM);
    __syncthreads();

    float ss = 0.f;
#pragma unroll
    for (int i = 0; i < 4; ++i) { float d = v[i] - mean; ss += d * d; }
#pragma unroll
    for (int m = 16; m >= 1; m >>= 1) ss += __shfl_xor(ss, m, 32);
    if ((threadIdx.x & 31) == 0) red[threadIdx.x >> 5] = ss;
    __syncthreads();
    float vt = 0.f;
#pragma unroll
    for (int i = 0; i < 8; ++i) vt += red[i];
    const float inv = 1.0f / (sqrtf(vt * (1.0f / (DIM - 1))) + EPS);

#pragma unroll
    for (int i = 0; i < 4; ++i) {
        const int c = threadIdx.x + i * 256;
        out[(size_t)row * DIM + c] = f2bf(gamma[c] * (v[i] - mean) * inv + beta[c]);
    }
}

// ---------------------------------------------------------------------------
// GEMM: C[M,N] = A[M,K](bf16, row-major) * Bt[N][K](bf16, pre-transposed)
// 128x128x64 tile, 8 waves, each wave 32x64 (2x4 WMMA accumulators).
// Register double-buffered global->LDS staging; epilogue fully templated.
// ---------------------------------------------------------------------------
#define LDK 72     // LDS pitch for 64-wide K tiles (+8 pad: conflict-free)

template <bool HAS_BIAS, bool RELU, bool HAS_RES, bool OUT_F32>
__global__ void __launch_bounds__(256)
gemm_kernel(const unsigned short* __restrict__ A,
            const unsigned short* __restrict__ Bt,
            const float* __restrict__ bias, const float* __restrict__ residual,
            unsigned short* __restrict__ outBf, float* __restrict__ outF,
            int M, int N, int K) {
    __shared__ unsigned short ldsA[128 * LDK];
    __shared__ unsigned short ldsB[128 * LDK];
    const int tid = threadIdx.x;
    const int lane = tid & 31, wave = tid >> 5;
    const int wm = wave >> 1, wn = wave & 1;            // 4x2 wave grid
    const int m0 = blockIdx.y * 128, n0 = blockIdx.x * 128;

    v8f acc[2][4];
#pragma unroll
    for (int i = 0; i < 2; ++i)
#pragma unroll
        for (int j = 0; j < 4; ++j)
#pragma unroll
            for (int v = 0; v < 8; ++v) acc[i][j][v] = 0.f;

    // staging: each thread copies 64B of A and 64B of Bt per K-step (BK=64)
    const int rowT = tid >> 1, colT = (tid & 1) * 32;
    const unsigned short* aptr = A  + (size_t)(m0 + rowT) * K + colT;
    const unsigned short* bptr = Bt + (size_t)(n0 + rowT) * K + colT;
    uint4* ldsAdst = (uint4*)(ldsA + rowT * LDK + colT);
    uint4* ldsBdst = (uint4*)(ldsB + rowT * LDK + colT);

    uint4 ra[4], rb[4];
#pragma unroll
    for (int i = 0; i < 4; ++i) {
        ra[i] = ((const uint4*)aptr)[i];
        rb[i] = ((const uint4*)bptr)[i];
    }

    for (int k0 = 0; k0 < K; k0 += 64) {
#pragma unroll
        for (int i = 0; i < 4; ++i) { ldsAdst[i] = ra[i]; ldsBdst[i] = rb[i]; }
        __syncthreads();

        if (k0 + 64 < K) {                              // prefetch next K-step
            aptr += 64; bptr += 64;
#pragma unroll
            for (int i = 0; i < 4; ++i) {
                ra[i] = ((const uint4*)aptr)[i];
                rb[i] = ((const uint4*)bptr)[i];
            }
        }

#pragma unroll
        for (int kk = 0; kk < 64; kk += 32) {
            v16bf a0 = load_a_frag(ldsA, wm * 32,      LDK, kk);
            v16bf a1 = load_a_frag(ldsA, wm * 32 + 16, LDK, kk);
#pragma unroll
            for (int nt = 0; nt < 4; ++nt) {
                v16bf bf = load_b_frag(ldsB, wn * 64 + nt * 16, LDK, kk);
                acc[0][nt] = wmma_bf16(a0, bf, acc[0][nt]);
                acc[1][nt] = wmma_bf16(a1, bf, acc[1][nt]);
            }
        }
        __syncthreads();
    }

    const int g = lane >> 4, l15 = lane & 15;
#pragma unroll
    for (int mt = 0; mt < 2; ++mt)
#pragma unroll
        for (int nt = 0; nt < 4; ++nt) {
            const int n = n0 + wn * 64 + nt * 16 + l15;
            const float bv = HAS_BIAS ? bias[n] : 0.0f;
#pragma unroll
            for (int v = 0; v < 8; ++v) {
                const int m = m0 + wm * 32 + mt * 16 + v + 8 * g;
                float val = acc[mt][nt][v] + bv;
                if (RELU) val = fmaxf(val, 0.0f);
                const size_t idx = (size_t)m * N + n;
                if (OUT_F32) {
                    float r = HAS_RES ? residual[idx] : 0.0f;
                    outF[idx] = r + val;
                } else {
                    outBf[idx] = f2bf(val);
                }
            }
        }
}

// ---------------------------------------------------------------------------
// Flash attention: grid (SEQ/128, BATCH*HEADS), 256 threads.
// Wave w owns q rows [qb*128 + w*16, +16), full head dim 64, online softmax.
// Row max via permlane16 (VALU); row sum via WMMA against all-ones B.
// ---------------------------------------------------------------------------
__global__ void __launch_bounds__(256)
attn_kernel(const unsigned short* __restrict__ qkv, unsigned short* __restrict__ out) {
    __shared__ unsigned short ldsK[32 * 72];      // K chunk  [key][d]
    __shared__ unsigned short ldsVt[64 * 40];     // V chunk transposed [d][key]
    __shared__ unsigned short ldsP[8 * 16 * 32];  // per-wave P transpose patch

    const int tid = threadIdx.x, lane = tid & 31, wave = tid >> 5;
    const int g = lane >> 4, l15 = lane & 15;
    const int qb = blockIdx.x, bh = blockIdx.y;
    const int b = bh >> 4, h = bh & 15;
    const size_t tok0 = (size_t)b * SEQ;
    const int q0 = qb * 128 + wave * 16;

    // all-ones bf16 B fragment for WMMA row sums
    union { v16bf v; unsigned int u[8]; } ones;
#pragma unroll
    for (int i = 0; i < 8; ++i) ones.u[i] = 0x3F803F80u;

    // Q fragments (2 x 16x32 over head dim 64), loaded once from global (bf16)
    v16bf qa[2];
#pragma unroll
    for (int f = 0; f < 2; ++f)
        qa[f] = load_a_frag(qkv + (tok0 + q0) * (3 * DIM) + h * HDIM, 0, 3 * DIM, f * 32);

    float mrun[8];            // running UNSCALED S max per owned row
    v8f lacc;                 // running softmax denominator (WMMA-accumulated)
    v8f acc[4];
#pragma unroll
    for (int v = 0; v < 8; ++v) { mrun[v] = -1e30f; lacc[v] = 0.f; }
#pragma unroll
    for (int d = 0; d < 4; ++d)
#pragma unroll
        for (int v = 0; v < 8; ++v) acc[d][v] = 0.f;

    for (int kc = 0; kc < SEQ; kc += 32) {
        // Stage K (row-major) and V (transposed), 32 keys x 64 dims, bf16
        {
            const int key = tid >> 3, dseg = (tid & 7) * 8;
            const unsigned short* src = qkv + (tok0 + kc + key) * (3 * DIM) + h * HDIM;
            uint4 kd = *(const uint4*)(src + DIM + dseg);        // K part
            *(uint4*)(ldsK + key * 72 + dseg) = kd;
            uint4 vd = *(const uint4*)(src + 2 * DIM + dseg);    // V part
            const unsigned short* vp = (const unsigned short*)&vd;
#pragma unroll
            for (int j = 0; j < 8; ++j) ldsVt[(dseg + j) * 40 + key] = vp[j];
        }
        __syncthreads();

        // S = Q * K^T  (two 16x16 key groups, K-dim = head dim in 2 chunks)
        v8f s0, s1;
#pragma unroll
        for (int v = 0; v < 8; ++v) { s0[v] = 0.f; s1[v] = 0.f; }
#pragma unroll
        for (int dc = 0; dc < 2; ++dc) {
            v16bf bk0 = load_b_frag(ldsK, 0,  72, dc * 32);
            v16bf bk1 = load_b_frag(ldsK, 16, 72, dc * 32);
            s0 = wmma_bf16(qa[dc], bk0, s0);
            s1 = wmma_bf16(qa[dc], bk1, s1);
        }

        // online softmax on UNSCALED S (scale folded into exp2 argument):
        // p = 2^((s - m) * SM_SCALE*log2(e)) ; row stats at row = v + 8*(lane/16)
        float corr[8];
#pragma unroll
        for (int v = 0; v < 8; ++v) {
            const float mv = grp16_max(fmaxf(s0[v], s1[v]));
            const float nm = fmaxf(mrun[v], mv);
            corr[v] = exp2f((mrun[v] - nm) * SM_SCALE_LOG2E);
            mrun[v] = nm;
        }

        unsigned short* pw = ldsP + wave * 512;
#pragma unroll
        for (int v = 0; v < 8; ++v) {
            const float p0 = exp2f((s0[v] - mrun[v]) * SM_SCALE_LOG2E);
            const float p1 = exp2f((s1[v] - mrun[v]) * SM_SCALE_LOG2E);
            const int r = v + 8 * g;                 // C-layout row
            pw[r * 32 + l15]      = f2bf(p0);        // keys 0..15
            pw[r * 32 + 16 + l15] = f2bf(p1);        // keys 16..31
            lacc[v] *= corr[v];
#pragma unroll
            for (int d = 0; d < 4; ++d) acc[d][v] *= corr[v];
        }

        // O += P * V ; l += P * ones   (A = P via LDS transpose patch)
        v16bf pa = load_a_frag(pw, 0, 32, 0);
        lacc = wmma_bf16(pa, ones.v, lacc);
#pragma unroll
        for (int dt = 0; dt < 4; ++dt) {
            v16bf bv = load_b_frag(ldsVt, dt * 16, 40, 0);
            acc[dt] = wmma_bf16(pa, bv, acc[dt]);
        }
        __syncthreads();
    }

    // epilogue: O /= l, store bf16 at [token][h*64 + d]
#pragma unroll
    for (int v = 0; v < 8; ++v) {
        const float inv = 1.0f / lacc[v];
        const size_t row = (tok0 + q0 + v + 8 * g) * (size_t)DIM + h * HDIM;
#pragma unroll
        for (int dt = 0; dt < 4; ++dt)
            out[row + dt * 16 + l15] = f2bf(acc[dt][v] * inv);
    }
}

// ---------------------------------------------------------------------------
extern "C" void kernel_launch(void* const* d_in, const int* in_sizes, int n_in,
                              void* d_out, int out_size, void* d_ws, size_t ws_size,
                              hipStream_t stream) {
    const float* x      = (const float*)d_in[0];
    const float* w_qkv  = (const float*)d_in[1];
    const float* w_out  = (const float*)d_in[2];
    const float* b_out  = (const float*)d_in[3];
    const float* w1     = (const float*)d_in[4];
    const float* b1     = (const float*)d_in[5];
    const float* w2     = (const float*)d_in[6];
    const float* b2     = (const float*)d_in[7];
    const float* gamma1 = (const float*)d_in[8];
    const float* beta1  = (const float*)d_in[9];
    const float* gamma2 = (const float*)d_in[10];
    const float* beta2  = (const float*)d_in[11];
    float* out = (float*)d_out;

    // workspace layout (bf16 activations; FFN hidden reuses dead qkv+attn region)
    unsigned short* xn    = (unsigned short*)d_ws;                // 8192x1024 bf16
    unsigned short* qkv   = xn   + (size_t)MTOT * DIM;            // 8192x3072 bf16
    unsigned short* attn  = qkv  + (size_t)MTOT * 3 * DIM;        // 8192x1024 bf16
    unsigned short* ffn1  = qkv;                                  // 8192x4096 bf16 (reuse)
    float*          x1    = (float*)(attn + (size_t)MTOT * DIM);  // 8192x1024 f32
    unsigned short* wtqkv = (unsigned short*)(x1 + (size_t)MTOT * DIM); // [3072][1024]
    unsigned short* wtout = wtqkv + (size_t)(3 * DIM) * DIM;      // [1024][1024]
    unsigned short* wt1   = wtout + (size_t)DIM * DIM;            // [4096][1024]
    unsigned short* wt2   = wt1   + (size_t)HIDDEN * DIM;         // [1024][4096]

    // 0) one-time weight convert+transpose (fp32 [K][N] -> bf16 [N][K])
    wt_kernel<<<dim3(3 * DIM / 64, DIM / 64), 256, 0, stream>>>(w_qkv, wtqkv, DIM, 3 * DIM);
    wt_kernel<<<dim3(DIM / 64, DIM / 64), 256, 0, stream>>>(w_out, wtout, DIM, DIM);
    wt_kernel<<<dim3(HIDDEN / 64, DIM / 64), 256, 0, stream>>>(w1, wt1, DIM, HIDDEN);
    wt_kernel<<<dim3(DIM / 64, HIDDEN / 64), 256, 0, stream>>>(w2, wt2, HIDDEN, DIM);

    // 1) h = norm(x)
    ln_kernel<<<MTOT, 256, 0, stream>>>(x, gamma1, beta1, xn);
    // 2) qkv = h @ w_qkv
    gemm_kernel<false, false, false, false>
        <<<dim3(3 * DIM / 128, MTOT / 128), 256, 0, stream>>>(
        xn, wtqkv, nullptr, nullptr, qkv, nullptr, MTOT, 3 * DIM, DIM);
    // 3) attention (softmax(q k^T / sqrt(D)) v)
    attn_kernel<<<dim3(SEQ / 128, BATCH * HEADS), 256, 0, stream>>>(qkv, attn);
    // 4) x1 = x + attn @ w_out + b_out
    gemm_kernel<true, false, true, true>
        <<<dim3(DIM / 128, MTOT / 128), 256, 0, stream>>>(
        attn, wtout, b_out, x, nullptr, x1, MTOT, DIM, DIM);
    // 5) h2 = norm(x1)
    ln_kernel<<<MTOT, 256, 0, stream>>>(x1, gamma2, beta2, xn);
    // 6) f = relu(h2 @ w1 + b1)
    gemm_kernel<true, true, false, false>
        <<<dim3(HIDDEN / 128, MTOT / 128), 256, 0, stream>>>(
        xn, wt1, b1, nullptr, ffn1, nullptr, MTOT, HIDDEN, DIM);
    // 7) out = x1 + f @ w2 + b2
    gemm_kernel<true, false, true, true>
        <<<dim3(DIM / 128, MTOT / 128), 256, 0, stream>>>(
        ffn1, wt2, b2, x1, nullptr, out, MTOT, DIM, HIDDEN);
}
